// PointNet2_model_29738353557549
// MI455X (gfx1250) — matrix operand order repaired and checked by reading, not compiled, not run on hardware
//
#include <hip/hip_runtime.h>

// ---------------------------------------------------------------------------
// Types for CDNA5 WMMA
// ---------------------------------------------------------------------------
typedef __attribute__((ext_vector_type(16))) _Float16 v16h;
typedef __attribute__((ext_vector_type(8)))  float    v8f;

#define LDT 40  // LDS row stride in f16 (80 bytes: 16B aligned, bank-skewed)

// ---------------------------------------------------------------------------
// GEMM + folded-BN + ReLU.  out[M,N] = act( A[M,K] * Wt[N,K]^T * scale + bias )
// M % 128 == 0, N % 64 == 0, K % 32 == 0 guaranteed by construction.
// Block: 256 threads = 8 waves; block tile 128x64; each wave computes a 32x32
// register tile (2x2 of 16x16) -> 4 v_wmma per 32-K chunk with 8 ds_load_b128
// (2 A frags + 2 B frags), i.e. 2 LDS loads per WMMA.
// ---------------------------------------------------------------------------
template <typename OutT>
__global__ __launch_bounds__(256) void gemm_bn_act(
    const _Float16* __restrict__ A, const _Float16* __restrict__ Wt,
    const float* __restrict__ scale, const float* __restrict__ bias,
    OutT* __restrict__ out, int M, int Ncols, int K, int relu) {
  __shared__ _Float16 As[128 * LDT];
  __shared__ _Float16 Bs[64 * LDT];

  const int tid  = threadIdx.x;
  const int wave = tid >> 5;
  const int lane = tid & 31;
  const int hl   = lane >> 4;   // lane half (0/1)
  const int l16  = lane & 15;
  const int m0   = blockIdx.y * 128;
  const int n0   = blockIdx.x * 64;
  const int wr   = (wave >> 1) * 32;  // wave row offset in tile (0,32,64,96)
  const int wc   = (wave & 1) * 32;   // wave col offset (0,32)

  v8f acc00 = {}, acc01 = {}, acc10 = {}, acc11 = {};

  const int srow = tid >> 2;        // 0..63
  const int scol = (tid & 3) * 8;   // 0,8,16,24 (f16 elems)

  for (int k0 = 0; k0 < K; k0 += 32) {
    // ---- stage 128x32 A tile (2 float4/thread) + 64x32 B tile into LDS ----
    *(float4*)&As[srow * LDT + scol] =
        *(const float4*)&A[(size_t)(m0 + srow) * K + k0 + scol];
    *(float4*)&As[(srow + 64) * LDT + scol] =
        *(const float4*)&A[(size_t)(m0 + srow + 64) * K + k0 + scol];
    *(float4*)&Bs[srow * LDT + scol] =
        *(const float4*)&Wt[(size_t)(n0 + srow) * K + k0 + scol];
    if (k0 + 32 < K)
      __builtin_prefetch(&A[(size_t)(m0 + srow) * K + k0 + 32 + scol], 0, 1);
    __syncthreads();

    // ---- assemble WMMA fragments per ISA 7.12.2 (16-bit A 16x32 layout) ----
    union Frag { v16h v; unsigned u[8]; } fa0, fa1, fb0, fb1;
    const _Float16* ar0 = &As[(wr + l16) * LDT];
    const _Float16* ar1 = &As[(wr + 16 + l16) * LDT];
    const _Float16* br0 = &Bs[(wc + l16) * LDT];
    const _Float16* br1 = &Bs[(wc + 16 + l16) * LDT];
#pragma unroll
    for (int vi = 0; vi < 8; ++vi) {
      const int ka = ((vi >> 2) << 4) + (hl << 3) + ((vi & 3) << 1);
      fa0.u[vi] = *(const unsigned*)(ar0 + ka);
      fa1.u[vi] = *(const unsigned*)(ar1 + ka);
    }
#pragma unroll
    for (int vi = 0; vi < 8; ++vi) {
      const int kb = ((vi >> 2) << 4) + (hl << 3) + ((vi & 3) << 1);
      fb0.u[vi] = *(const unsigned*)(br0 + kb);
      fb1.u[vi] = *(const unsigned*)(br1 + kb);
    }
    acc00 = __builtin_amdgcn_wmma_f32_16x16x32_f16(false, fa0.v, false, fb0.v,
                                                   (short)0, acc00, false, false);
    acc01 = __builtin_amdgcn_wmma_f32_16x16x32_f16(false, fa0.v, false, fb1.v,
                                                   (short)0, acc01, false, false);
    acc10 = __builtin_amdgcn_wmma_f32_16x16x32_f16(false, fa1.v, false, fb0.v,
                                                   (short)0, acc10, false, false);
    acc11 = __builtin_amdgcn_wmma_f32_16x16x32_f16(false, fa1.v, false, fb1.v,
                                                   (short)0, acc11, false, false);
    __syncthreads();
  }

  // ---- epilogue: folded BN (scale,bias) + ReLU, C/D layout m=r+8*hl -------
  const int nc0 = n0 + wc + l16;
  const int nc1 = nc0 + 16;
  const float s0 = scale[nc0], b0 = bias[nc0];
  const float s1 = scale[nc1], b1 = bias[nc1];
#pragma unroll
  for (int r = 0; r < 8; ++r) {
    const int ma = m0 + wr + hl * 8 + r;
    const int mb = ma + 16;
    float y00 = acc00[r] * s0 + b0;
    float y01 = acc01[r] * s1 + b1;
    float y10 = acc10[r] * s0 + b0;
    float y11 = acc11[r] * s1 + b1;
    if (relu) {
      y00 = fmaxf(y00, 0.f); y01 = fmaxf(y01, 0.f);
      y10 = fmaxf(y10, 0.f); y11 = fmaxf(y11, 0.f);
    }
    out[(size_t)ma * Ncols + nc0] = (OutT)y00;
    out[(size_t)ma * Ncols + nc1] = (OutT)y01;
    out[(size_t)mb * Ncols + nc0] = (OutT)y10;
    out[(size_t)mb * Ncols + nc1] = (OutT)y11;
  }
}

// ---------------------------------------------------------------------------
// Weight prep: Wt[n][kpad] = f16(W[k][n]) (transposed, K zero-padded) and BN
// fold: scale = gamma*rsqrt(var+eps); bias = beta + (b-mean)*scale.
// ---------------------------------------------------------------------------
__global__ void prep_weights(const float* __restrict__ W, const float* __restrict__ b,
                             const float* __restrict__ gamma, const float* __restrict__ beta,
                             const float* __restrict__ mean, const float* __restrict__ var,
                             _Float16* __restrict__ Wt, float* __restrict__ scale,
                             float* __restrict__ bias, int K, int Kpad, int Nout) {
  int g = blockIdx.x * blockDim.x + threadIdx.x;
  if (g >= Nout * Kpad) return;
  int n = g / Kpad, k = g % Kpad;
  Wt[(size_t)n * Kpad + k] = (k < K) ? (_Float16)W[(size_t)k * Nout + n] : (_Float16)0.f;
  if (k == 0) {
    float s = gamma[n] * rsqrtf(var[n] + 1e-5f);
    scale[n] = s;
    bias[n]  = beta[n] + (b[n] - mean[n]) * s;
  }
}

// ---------------------------------------------------------------------------
// Farthest point sampling: one block per batch, dist[] in LDS, argmax tree.
// ---------------------------------------------------------------------------
__global__ __launch_bounds__(256) void fps_kernel(
    const float* __restrict__ xyz, int Npts, int S,
    float* __restrict__ new_xyz, int* __restrict__ fps_idx) {
  __shared__ float dist[4096];
  __shared__ float rv[256];
  __shared__ int   ri[256];
  __shared__ int   sfar;
  const int b = blockIdx.x, tid = threadIdx.x;
  const float* px = xyz + (size_t)b * Npts * 3;
  for (int i = tid; i < Npts; i += blockDim.x) dist[i] = 1e10f;
  if (tid == 0) sfar = 0;
  __syncthreads();
  for (int it = 0; it < S; ++it) {
    const int far = sfar;
    if (tid == 0) {
      fps_idx[b * S + it] = far;
      new_xyz[(size_t)(b * S + it) * 3 + 0] = px[far * 3 + 0];
      new_xyz[(size_t)(b * S + it) * 3 + 1] = px[far * 3 + 1];
      new_xyz[(size_t)(b * S + it) * 3 + 2] = px[far * 3 + 2];
    }
    const float cx = px[far * 3 + 0], cy = px[far * 3 + 1], cz = px[far * 3 + 2];
    float best = -1.f; int bn = 0;
    for (int n = tid; n < Npts; n += blockDim.x) {
      float dx = px[n * 3 + 0] - cx, dy = px[n * 3 + 1] - cy, dz = px[n * 3 + 2] - cz;
      float nd = fminf(dist[n], dx * dx + dy * dy + dz * dz);
      dist[n] = nd;
      if (nd > best) { best = nd; bn = n; }
    }
    rv[tid] = best; ri[tid] = bn;
    __syncthreads();
    for (int s = 128; s > 0; s >>= 1) {
      if (tid < s && rv[tid + s] > rv[tid]) { rv[tid] = rv[tid + s]; ri[tid] = ri[tid + s]; }
      __syncthreads();
    }
    if (tid == 0) sfar = ri[0];
    __syncthreads();
  }
}

// ---------------------------------------------------------------------------
// Ball query: first K point indices (ascending) within radius^2 of centroid,
// padded with the first hit (matches reference's sort-then-replace trick).
// ---------------------------------------------------------------------------
__global__ void ball_query_kernel(const float* __restrict__ xyz,
                                  const float* __restrict__ new_xyz,
                                  int* __restrict__ idx, int Npts, int S, int K,
                                  float r2, int total) {
  int g = blockIdx.x * blockDim.x + threadIdx.x;
  if (g >= total) return;
  int b = g / S;
  const float* p = xyz + (size_t)b * Npts * 3;
  const float cx = new_xyz[(size_t)g * 3 + 0];
  const float cy = new_xyz[(size_t)g * 3 + 1];
  const float cz = new_xyz[(size_t)g * 3 + 2];
  int* o = idx + (size_t)g * K;
  int cnt = 0, first = 0; bool has = false;
  for (int n = 0; n < Npts && cnt < K; ++n) {
    float dx = p[n * 3 + 0] - cx, dy = p[n * 3 + 1] - cy, dz = p[n * 3 + 2] - cz;
    if (dx * dx + dy * dy + dz * dz <= r2) {
      if (!has) { first = n; has = true; }
      o[cnt++] = n;
    }
  }
  for (; cnt < K; ++cnt) o[cnt] = first;
}

// SA1 grouping: [rel_xyz(3) | desc(Cdesc from [B,C,N] layout)] -> f16, K-padded
__global__ void group_sa1(const float* __restrict__ xyz, const float* __restrict__ desc,
                          const float* __restrict__ new_xyz, const int* __restrict__ idx,
                          _Float16* __restrict__ out, int Npts, int S, int K, int Cdesc,
                          int Kpad, int total) {
  int g = blockIdx.x * blockDim.x + threadIdx.x;
  if (g >= total) return;
  int row = g / Kpad, c = g % Kpad;
  int b = row / (S * K), s = (row / K) % S, k = row % K;
  int n = idx[((size_t)b * S + s) * K + k];
  float v = 0.f;
  if (c < 3)
    v = xyz[((size_t)b * Npts + n) * 3 + c] - new_xyz[((size_t)b * S + s) * 3 + c];
  else if (c < 3 + Cdesc)
    v = desc[((size_t)b * Cdesc + (c - 3)) * Npts + n];
  out[(size_t)row * Kpad + c] = (_Float16)v;
}

// SA2 grouping: features come from f16 pooled buffer [B*Npts][C]
__global__ void group_sa2(const float* __restrict__ xyz, const _Float16* __restrict__ feat,
                          const float* __restrict__ new_xyz, const int* __restrict__ idx,
                          _Float16* __restrict__ out, int Npts, int S, int K, int C,
                          int Kpad, int total) {
  int g = blockIdx.x * blockDim.x + threadIdx.x;
  if (g >= total) return;
  int row = g / Kpad, c = g % Kpad;
  int b = row / (S * K), s = (row / K) % S, k = row % K;
  int n = idx[((size_t)b * S + s) * K + k];
  float v = 0.f;
  if (c < 3)
    v = xyz[((size_t)b * Npts + n) * 3 + c] - new_xyz[((size_t)b * S + s) * 3 + c];
  else if (c < 3 + C)
    v = (float)feat[((size_t)b * Npts + n) * C + (c - 3)];
  out[(size_t)row * Kpad + c] = (_Float16)v;
}

// SA3 "group all": [xyz(3) | feat(C)] per point, K-padded
__global__ void group_all(const float* __restrict__ xyz, const _Float16* __restrict__ feat,
                          _Float16* __restrict__ out, int C, int Kpad, int total) {
  int g = blockIdx.x * blockDim.x + threadIdx.x;
  if (g >= total) return;
  int row = g / Kpad, c = g % Kpad;
  float v = 0.f;
  if (c < 3) v = xyz[(size_t)row * 3 + c];
  else if (c < 3 + C) v = (float)feat[(size_t)row * C + (c - 3)];
  out[(size_t)row * Kpad + c] = (_Float16)v;
}

// Max-pool over group dim: in [R*K][C] -> out [R][C]
__global__ void maxpool_kernel(const _Float16* __restrict__ in, _Float16* __restrict__ out,
                               int R, int K, int C, int total) {
  int g = blockIdx.x * blockDim.x + threadIdx.x;
  if (g >= total) return;
  int r = g / C, c = g % C;
  float m = -1e30f;
  for (int k = 0; k < K; ++k) m = fmaxf(m, (float)in[((size_t)r * K + k) * C + c]);
  out[(size_t)r * C + c] = (_Float16)m;
}

__global__ void fill_zero_h(_Float16* p, int n) {
  int i = blockIdx.x * blockDim.x + threadIdx.x;
  if (i < n) p[i] = (_Float16)0.f;
}

// fc3 input: rows B*N, cols 512 = [desc[b,:,n] | headfeat[b,:]]
__global__ void build_fc3(const float* __restrict__ desc, const _Float16* __restrict__ x2,
                          _Float16* __restrict__ out, int Npts, int total) {
  int g = blockIdx.x * blockDim.x + threadIdx.x;
  if (g >= total) return;
  int row = g >> 9, c = g & 511;          // 512 cols
  int b = row / Npts, n = row % Npts;
  float v = (c < 256) ? desc[((size_t)b * 256 + c) * Npts + n]
                      : (float)x2[(size_t)b * 256 + (c - 256)];
  out[(size_t)row * 512 + c] = (_Float16)v;
}

// final: rowmajor [B*N][256] f32 -> d_out [B][256][N]
__global__ void final_transpose(const float* __restrict__ in, float* __restrict__ out,
                                int Npts, int total) {
  int g = blockIdx.x * blockDim.x + threadIdx.x;
  if (g >= total) return;
  int row = g >> 8, c = g & 255;
  int b = row / Npts, n = row % Npts;
  out[((size_t)b * 256 + c) * Npts + n] = in[(size_t)row * 256 + c];
}

// ---------------------------------------------------------------------------
// Host-side orchestration
// ---------------------------------------------------------------------------
struct LayerMeta { int K, Kpad, N, base; };

extern "C" void kernel_launch(void* const* d_in, const int* in_sizes, int n_in,
                              void* d_out, int out_size, void* d_ws, size_t ws_size,
                              hipStream_t stream) {
  (void)in_sizes; (void)n_in; (void)out_size; (void)ws_size;
  constexpr int B = 8, N1 = 4096, S1 = 512, K1 = 32, S2 = 128, K2 = 64;
  constexpr int MH = 128;  // head batch rows padded to GEMM M-tile (128)

  static const LayerMeta L[12] = {
      {259, 288, 256, 2},   {256, 256, 256, 8},   {256, 256, 512, 14},   // SA1
      {515, 544, 512, 20},  {512, 512, 512, 26},  {512, 512, 1024, 32},  // SA2
      {1027, 1056, 1024, 38}, {1024, 1024, 2048, 44}, {2048, 2048, 2048, 50}, // SA3
      {2048, 2048, 1024, 56}, {1024, 1024, 256, 62}, {512, 512, 256, 68}};    // head

  // ---- bump allocator over workspace ----
  char* ws = (char*)d_ws;
  size_t off = 0;
  auto alloc = [&](size_t bytes) -> void* {
    void* p = ws + off;
    off = (off + bytes + 255) & ~(size_t)255;
    return p;
  };

  _Float16* Wt[12]; float* sc[12]; float* bi[12];
  for (int i = 0; i < 12; ++i) {
    Wt[i] = (_Float16*)alloc((size_t)L[i].N * L[i].Kpad * 2);
    sc[i] = (float*)alloc((size_t)L[i].N * 4);
    bi[i] = (float*)alloc((size_t)L[i].N * 4);
  }
  _Float16* bufA   = (_Float16*)alloc((size_t)131072 * 288 * 2); // 75.5 MB
  _Float16* bufB   = (_Float16*)alloc((size_t)131072 * 512 * 2); // 134 MB
  float* new_xyz1  = (float*)alloc((size_t)B * S1 * 3 * 4);
  float* new_xyz2  = (float*)alloc((size_t)B * S2 * 3 * 4);
  int*   fps_idx1  = (int*)alloc((size_t)B * S1 * 4);
  int*   fps_idx2  = (int*)alloc((size_t)B * S2 * 4);
  int*   idx1      = (int*)alloc((size_t)B * S1 * K1 * 4);
  int*   idx2      = (int*)alloc((size_t)B * S2 * K2 * 4);
  _Float16* l1feat = (_Float16*)alloc((size_t)B * S1 * 512 * 2);
  _Float16* l2feat = (_Float16*)alloc((size_t)B * S2 * 1024 * 2);
  _Float16* headbf = (_Float16*)alloc((size_t)MH * 2048 * 2);
  _Float16* fc1out = (_Float16*)alloc((size_t)MH * 1024 * 2);
  _Float16* x2buf  = (_Float16*)alloc((size_t)MH * 256 * 2);

  const float* xyz  = (const float*)d_in[0];
  const float* desc = (const float*)d_in[1];

  // ---- weight prep (12 layers) ----
  for (int i = 0; i < 12; ++i) {
    int total = L[i].N * L[i].Kpad;
    prep_weights<<<(total + 255) / 256, 256, 0, stream>>>(
        (const float*)d_in[L[i].base + 0], (const float*)d_in[L[i].base + 1],
        (const float*)d_in[L[i].base + 2], (const float*)d_in[L[i].base + 3],
        (const float*)d_in[L[i].base + 4], (const float*)d_in[L[i].base + 5],
        Wt[i], sc[i], bi[i], L[i].K, L[i].Kpad, L[i].N);
  }

  auto gemm_h = [&](const _Float16* A, int li, _Float16* out, int M) {
    dim3 g(L[li].N / 64, M / 128);
    gemm_bn_act<_Float16><<<g, 256, 0, stream>>>(A, Wt[li], sc[li], bi[li], out,
                                                 M, L[li].N, L[li].Kpad, 1);
  };

  // ================= SA1 =================
  fps_kernel<<<B, 256, 0, stream>>>(xyz, N1, S1, new_xyz1, fps_idx1);
  {
    int tot = B * S1;
    ball_query_kernel<<<(tot + 255) / 256, 256, 0, stream>>>(xyz, new_xyz1, idx1,
                                                             N1, S1, K1, 0.04f, tot);
  }
  {
    int tot = B * S1 * K1 * 288;
    group_sa1<<<(tot + 255) / 256, 256, 0, stream>>>(xyz, desc, new_xyz1, idx1,
                                                     bufA, N1, S1, K1, 256, 288, tot);
  }
  gemm_h(bufA, 0, bufB, B * S1 * K1);   // 288 -> 256
  gemm_h(bufB, 1, bufA, B * S1 * K1);   // 256 -> 256
  gemm_h(bufA, 2, bufB, B * S1 * K1);   // 256 -> 512
  {
    int tot = B * S1 * 512;
    maxpool_kernel<<<(tot + 255) / 256, 256, 0, stream>>>(bufB, l1feat, B * S1, K1, 512, tot);
  }

  // ================= SA2 =================
  fps_kernel<<<B, 256, 0, stream>>>(new_xyz1, S1, S2, new_xyz2, fps_idx2);
  {
    int tot = B * S2;
    ball_query_kernel<<<(tot + 255) / 256, 256, 0, stream>>>(new_xyz1, new_xyz2, idx2,
                                                             S1, S2, K2, 0.16f, tot);
  }
  {
    int tot = B * S2 * K2 * 544;
    group_sa2<<<(tot + 255) / 256, 256, 0, stream>>>(new_xyz1, l1feat, new_xyz2, idx2,
                                                     bufA, S1, S2, K2, 512, 544, tot);
  }
  gemm_h(bufA, 3, bufB, B * S2 * K2);   // 544 -> 512
  gemm_h(bufB, 4, bufA, B * S2 * K2);   // 512 -> 512
  gemm_h(bufA, 5, bufB, B * S2 * K2);   // 512 -> 1024
  {
    int tot = B * S2 * 1024;
    maxpool_kernel<<<(tot + 255) / 256, 256, 0, stream>>>(bufB, l2feat, B * S2, K2, 1024, tot);
  }

  // ================= SA3 (group all) =================
  {
    int tot = B * S2 * 1056;
    group_all<<<(tot + 255) / 256, 256, 0, stream>>>(new_xyz2, l2feat, bufA, 1024, 1056, tot);
  }
  gemm_h(bufA, 6, bufB, B * S2);        // 1056 -> 1024  (M=1024)
  gemm_h(bufB, 7, bufA, B * S2);        // 1024 -> 2048
  gemm_h(bufA, 8, bufB, B * S2);        // 2048 -> 2048
  fill_zero_h<<<(MH * 2048 + 255) / 256, 256, 0, stream>>>(headbf, MH * 2048);
  {
    int tot = B * 2048;
    maxpool_kernel<<<(tot + 255) / 256, 256, 0, stream>>>(bufB, headbf, B, S2, 2048, tot);
  }

  // ================= head =================
  gemm_h(headbf, 9, fc1out, MH);        // fc1: 2048 -> 1024  (M padded to 128)
  gemm_h(fc1out, 10, x2buf, MH);        // fc2: 1024 -> 256
  {
    int tot = B * N1 * 512;
    build_fc3<<<(tot + 255) / 256, 256, 0, stream>>>(desc, x2buf, bufB, N1, tot);
  }
  {
    dim3 g(L[11].N / 64, (B * N1) / 128);
    gemm_bn_act<float><<<g, 256, 0, stream>>>(bufB, Wt[11], sc[11], bi[11],
                                              (float*)bufA, B * N1, 256, 512, 1);
  }
  {
    int tot = B * N1 * 256;
    final_transpose<<<(tot + 255) / 256, 256, 0, stream>>>((const float*)bufA,
                                                           (float*)d_out, N1, tot);
  }
}
